// RTGNTorsionMemory_74543452389456
// MI455X (gfx1250) — matrix operand order repaired and checked by reading, not compiled, not run on hardware
//
#include <hip/hip_runtime.h>
#include <hip/hip_bf16.h>
#include <math.h>

namespace {

constexpr int kDim   = 64;
constexpr int kNodes = 20000;
constexpr int kEdges = 40000;
constexpr int kTor   = 512;

typedef __bf16 bf16;
typedef __attribute__((ext_vector_type(16))) bf16 bf16x16;
typedef __attribute__((ext_vector_type(8)))  bf16 bf16x8;
typedef __attribute__((ext_vector_type(8)))  float f32x8;
typedef __attribute__((ext_vector_type(4)))  unsigned u32x4;

__device__ __forceinline__ float sigm(float x) { return 1.0f / (1.0f + __expf(-x)); }
__device__ __forceinline__ unsigned fenc(float f) {
  unsigned u = __float_as_uint(f);
  return (u & 0x80000000u) ? ~u : (u | 0x80000000u);
}
__device__ __forceinline__ float fdec(unsigned u) {
  return __uint_as_float((u & 0x80000000u) ? (u & 0x7fffffffu) : ~u);
}

// ------------------------------------------------------------------
// elementwise / conversion kernels
// ------------------------------------------------------------------
__global__ void k_f32_to_bf16(const float* __restrict__ src, bf16* __restrict__ dst, int n) {
  int i = blockIdx.x * blockDim.x + threadIdx.x;
  if (i < n) dst[i] = (bf16)src[i];
}

__global__ void k_transpose64_bf16(const float* __restrict__ src, bf16* __restrict__ dst) {
  int i = blockIdx.x * blockDim.x + threadIdx.x;  // 4096
  if (i < 64 * 64) {
    int k = i >> 6, n = i & 63;
    dst[n * 64 + k] = (bf16)src[i];                 // Bt[n,k] = conv_root[k,n]
  }
}

__global__ void k_vadd(const float* __restrict__ a, const float* __restrict__ b,
                       float* __restrict__ dst, int n) {
  int i = blockIdx.x * blockDim.x + threadIdx.x;
  if (i < n) dst[i] = a[i] + b[i];
}

__global__ void k_lin0(const float* __restrict__ x, const float* __restrict__ w,
                       const float* __restrict__ b, float* __restrict__ out) {
  int i = blockIdx.x * blockDim.x + threadIdx.x;
  if (i >= kNodes * kDim) return;
  int n = i >> 6, d = i & 63;
  float a = b[d] + x[n * 3] * w[d * 3] + x[n * 3 + 1] * w[d * 3 + 1] + x[n * 3 + 2] * w[d * 3 + 2];
  out[i] = fmaxf(a, 0.0f);
}

__global__ void k_edge_h1(const float* __restrict__ ea, const float* __restrict__ w1,
                          const float* __restrict__ b1, bf16* __restrict__ h1) {
  int i = blockIdx.x * blockDim.x + threadIdx.x;
  if (i >= kEdges * 128) return;
  int e = i >> 7, j = i & 127;
  float a = b1[j];
#pragma unroll
  for (int k = 0; k < 7; ++k) a = fmaf(ea[e * 7 + k], w1[j * 7 + k], a);
  h1[i] = (bf16)fmaxf(a, 0.0f);
}

__global__ void k_deg(const int* __restrict__ dst, float* __restrict__ deg, int e) {
  int i = blockIdx.x * blockDim.x + threadIdx.x;
  if (i < e) atomicAdd(&deg[dst[i]], 1.0f);
}

// ------------------------------------------------------------------
// Generic bf16 WMMA GEMM: C[M,N] = A[M,K] @ Bt[N,K]^T (+bias)(+relu)
// One wave per 16x16 tile, K in steps of 32 via v_wmma_f32_16x16x32_bf16.
// Fragment layouts per CDNA5 ISA 7.12.2 (16-bit A 16x32, B 32x16, f32 C 16x16).
// bf16 output path (ew tensor) uses non-temporal stores: the 327MB stream
// cannot live in the 192MB L2 anyway.
// ------------------------------------------------------------------
__device__ __forceinline__ bf16x16 load_fragA(const bf16* p) {
  bf16x8 lo = *reinterpret_cast<const bf16x8*>(p);        // K = kb+0..7
  bf16x8 hi = *reinterpret_cast<const bf16x8*>(p + 16);   // K = kb+16..23
  bf16x16 f;
#pragma unroll
  for (int i = 0; i < 8; ++i) { f[i] = lo[i]; f[i + 8] = hi[i]; }
  return f;
}
__device__ __forceinline__ bf16x16 load_fragB(const bf16* p) {
  bf16x8 lo = *reinterpret_cast<const bf16x8*>(p);        // K = kb+0..7
  bf16x8 hi = *reinterpret_cast<const bf16x8*>(p + 16);   // K = kb+16..23
  bf16x16 f;
#pragma unroll
  for (int i = 0; i < 8; ++i) { f[i] = lo[i]; f[i + 8] = hi[i]; }
  return f;
}

__global__ void k_wmma_gemm(const bf16* __restrict__ A, int lda,
                            const bf16* __restrict__ Bt, int ldb,
                            float* __restrict__ Cf, bf16* __restrict__ Cb, int ldc,
                            const float* __restrict__ bias,
                            int tilesM, int tilesN, int K, int relu) {
  int wave = blockIdx.x * (blockDim.x >> 5) + (threadIdx.x >> 5);
  int lane = threadIdx.x & 31;
  if (wave >= tilesM * tilesN) return;
  int tm = wave / tilesN, tn = wave % tilesN;
  const bf16* arow = A + (size_t)(tm * 16 + (lane & 15)) * lda + ((lane >> 4) * 8);
  const bf16* brow = Bt + (size_t)(tn * 16 + (lane & 15)) * ldb + ((lane >> 4) * 8);
  f32x8 acc = {};
  for (int k0 = 0; k0 < K; k0 += 32) {
    bf16x16 a = load_fragA(arow + k0);
    bf16x16 b = load_fragB(brow + k0);
    acc = __builtin_amdgcn_wmma_f32_16x16x32_bf16(false, a, false, b, (short)0, acc, false, false);
  }
  int cn = tn * 16 + (lane & 15);
  int rb = tm * 16 + ((lane >> 4) * 8);
  float bv = bias ? bias[cn] : 0.0f;
#pragma unroll
  for (int j = 0; j < 8; ++j) {
    float v = acc[j] + bv;
    if (relu) v = fmaxf(v, 0.0f);
    size_t idx = (size_t)(rb + j) * (size_t)ldc + (size_t)cn;
    if (Cf) Cf[idx] = v;
    if (Cb) __builtin_nontemporal_store((bf16)v, Cb + idx);
  }
}

// ------------------------------------------------------------------
// Per-edge matvec + scatter: msg[e] = out[src[e]] (1x64) @ ew[e] (64x64 bf16)
// One wave per edge. Lane l loads b128 (8 bf16 = one output-group chunk of
// row i0 + (l>>3)) -> 512B/wave per load instruction, non-temporal (327MB
// stream vs 192MB L2). Cross-lane 4-way shfl_xor reduction combines the 4
// row-subsets, then lanes 0..7 scatter 64 f32 atomics to the dst node.
// ------------------------------------------------------------------
__global__ void k_edge_msg(const float* __restrict__ out, const bf16* __restrict__ ews,
                           const int* __restrict__ src, const int* __restrict__ dst,
                           float* __restrict__ agg) {
  int e = blockIdx.x * (blockDim.x >> 5) + (threadIdx.x >> 5);
  int lane = threadIdx.x & 31;
  if (e >= kEdges) return;
  int s = src[e], d = dst[e];
  float x0 = out[s * 64 + 2 * lane];
  float x1 = out[s * 64 + 2 * lane + 1];
  int rsub = lane >> 3;   // which of 4 rows in each load group
  int og   = lane & 7;    // output group: outputs og*8 .. og*8+7
  const u32x4* W4 = reinterpret_cast<const u32x4*>(ews + (size_t)e * 4096);
  float acc[8] = {0.f, 0.f, 0.f, 0.f, 0.f, 0.f, 0.f, 0.f};
#pragma unroll 4
  for (int i0 = 0; i0 < 64; i0 += 4) {
    u32x4 w = __builtin_nontemporal_load(W4 + i0 * 8 + lane);  // row i0+rsub, cols og*8..+7
    int row = i0 + rsub;
    float xi = __shfl((row & 1) ? x1 : x0, row >> 1, 32);
#pragma unroll
    for (int q = 0; q < 4; ++q) {
      unsigned pq = w[q];
      acc[2 * q]     = fmaf(xi, __uint_as_float(pq << 16), acc[2 * q]);
      acc[2 * q + 1] = fmaf(xi, __uint_as_float(pq & 0xffff0000u), acc[2 * q + 1]);
    }
  }
#pragma unroll
  for (int j = 0; j < 8; ++j) {
    acc[j] += __shfl_xor(acc[j], 8, 32);
    acc[j] += __shfl_xor(acc[j], 16, 32);
  }
  if (lane < 8) {
#pragma unroll
    for (int j = 0; j < 8; ++j) atomicAdd(&agg[d * 64 + og * 8 + j], acc[j]);
  }
}

__global__ void k_m_point(const float* __restrict__ agg, const float* __restrict__ deg,
                          const float* __restrict__ root, const float* __restrict__ cbias,
                          bf16* __restrict__ mB) {
  int i = blockIdx.x * blockDim.x + threadIdx.x;
  if (i >= kNodes * kDim) return;
  int n = i >> 6, d = i & 63;
  float dg = fmaxf(deg[n], 1.0f);
  float v = agg[i] / dg + root[i] + cbias[d];
  mB[i] = (bf16)fmaxf(v, 0.0f);
}

__global__ void k_gru_point(const float* __restrict__ gi, const float* __restrict__ gh,
                            float* __restrict__ out) {
  int i = blockIdx.x * blockDim.x + threadIdx.x;
  if (i >= kNodes * kDim) return;
  int n = i >> 6, d = i & 63;
  const float* a = gi + (size_t)n * 192;
  const float* b = gh + (size_t)n * 192;
  float r  = sigm(a[d] + b[d]);
  float z  = sigm(a[64 + d] + b[64 + d]);
  float nn = tanhf(a[128 + d] + r * b[128 + d]);
  out[i] = (1.0f - z) * nn + z * out[i];
}

// ------------------------------------------------------------------
// Set2Set (single graph)
// ------------------------------------------------------------------
__global__ void k_s2s_lstm(const float* __restrict__ qstar, float* __restrict__ sh,
                           float* __restrict__ sc, const float* __restrict__ wih,
                           const float* __restrict__ whh, const float* __restrict__ bih,
                           const float* __restrict__ bhh) {
  __shared__ float sg[256];
  int t = threadIdx.x;  // 256 gate elements
  float g = bih[t] + bhh[t];
  for (int k = 0; k < 128; ++k) g = fmaf(qstar[k], wih[t * 128 + k], g);
  for (int k = 0; k < 64; ++k)  g = fmaf(sh[k], whh[t * 64 + k], g);
  sg[t] = g;
  __syncthreads();
  if (t < 64) {
    float gi = sg[t], gf = sg[64 + t], gg = sg[128 + t], go = sg[192 + t];
    float c2 = sigm(gf) * sc[t] + sigm(gi) * tanhf(gg);
    float h2 = sigm(go) * tanhf(c2);
    sc[t] = c2;
    sh[t] = h2;
  }
}

__global__ void k_s2s_init(float* __restrict__ scal, float* __restrict__ rraw) {
  int t = threadIdx.x;
  if (t < 64) rraw[t] = 0.0f;
  if (t == 0) {
    reinterpret_cast<unsigned*>(scal)[0] = 0x007fffffu;  // fenc(-inf)
    scal[1] = 0.0f;
  }
}

__global__ void k_dots(const float* __restrict__ out, const float* __restrict__ sh,
                       float* __restrict__ dots) {
  int n = blockIdx.x * (blockDim.x >> 5) + (threadIdx.x >> 5);
  int lane = threadIdx.x & 31;
  if (n >= kNodes) return;
  float v = out[n * 64 + 2 * lane] * sh[2 * lane] + out[n * 64 + 2 * lane + 1] * sh[2 * lane + 1];
#pragma unroll
  for (int o = 16; o > 0; o >>= 1) v += __shfl_down(v, o, 32);
  if (lane == 0) dots[n] = v;
}

__global__ void k_maxred(const float* __restrict__ dots, float* __restrict__ scal) {
  __shared__ float red[256];
  int tid = threadIdx.x;
  float m = -3.0e38f;
  for (int i = blockIdx.x * blockDim.x + tid; i < kNodes; i += gridDim.x * blockDim.x)
    m = fmaxf(m, dots[i]);
  red[tid] = m;
  __syncthreads();
  for (int s = 128; s > 0; s >>= 1) {
    if (tid < s) red[tid] = fmaxf(red[tid], red[tid + s]);
    __syncthreads();
  }
  if (tid == 0) atomicMax(reinterpret_cast<unsigned*>(scal), fenc(red[0]));
}

__global__ void k_attn(const float* __restrict__ dots, const float* __restrict__ out,
                       float* __restrict__ scal, float* __restrict__ rraw) {
  __shared__ float sr[64];
  __shared__ float ssum;
  int tid = threadIdx.x;
  if (tid < 64) sr[tid] = 0.0f;
  if (tid == 0) ssum = 0.0f;
  __syncthreads();
  int n = blockIdx.x * (blockDim.x >> 5) + (tid >> 5);
  int lane = tid & 31;
  if (n < kNodes) {
    float mx = fdec(reinterpret_cast<const unsigned*>(scal)[0]);
    float e = __expf(dots[n] - mx);
    atomicAdd(&sr[2 * lane], e * out[n * 64 + 2 * lane]);
    atomicAdd(&sr[2 * lane + 1], e * out[n * 64 + 2 * lane + 1]);
    if (lane == 0) atomicAdd(&ssum, e);
  }
  __syncthreads();
  if (tid < 64) atomicAdd(&rraw[tid], sr[tid]);
  if (tid == 0) atomicAdd(&scal[1], ssum);
}

__global__ void k_qstar(const float* __restrict__ rraw, const float* __restrict__ scal,
                        const float* __restrict__ sh, float* __restrict__ qstar) {
  int t = threadIdx.x;
  if (t < 64) {
    qstar[t] = sh[t];
    qstar[64 + t] = rraw[t] / scal[1];
  }
}

// ------------------------------------------------------------------
// Actor head
// ------------------------------------------------------------------
__global__ void k_build_z(const float* __restrict__ out, const int* __restrict__ nonring,
                          const float* __restrict__ qstar, bf16* __restrict__ zb) {
  int i = blockIdx.x * blockDim.x + threadIdx.x;
  if (i >= kTor * 384) return;
  int t = i / 384, c = i % 384;
  float v;
  if (c < 256) {
    int m = c * kTor + t;             // .view(4*dim, T).permute(1,0)
    int a = m >> 6, d = m & 63;
    v = out[(size_t)nonring[a] * 64 + d];
  } else {
    v = qstar[(t * 128 + (c - 256)) >> 9];  // repeat_interleave semantics
  }
  zb[i] = (bf16)v;
}

__global__ void k_lstm_act(const float* __restrict__ g, bf16* __restrict__ hb) {
  int i = blockIdx.x * blockDim.x + threadIdx.x;
  if (i >= kTor * 384) return;
  int t = i / 384, j = i % 384;
  const float* gr = g + (size_t)t * 1536;
  float gi = gr[j], gg = gr[768 + j], go = gr[1152 + j];   // c0 = 0 -> forget term vanishes
  float c2 = sigm(gi) * tanhf(gg);
  hb[i] = (bf16)(sigm(go) * tanhf(c2));
}

__global__ void k_logits_ent(const float* __restrict__ o, const float* __restrict__ w2,
                             const float* __restrict__ b2, float* __restrict__ dout) {
  __shared__ float sl[8];
  int t = blockIdx.x;          // torsion
  int tid = threadIdx.x;       // 64 threads
  if (tid < 6) {
    float a = b2[tid];
    for (int k = 0; k < 128; ++k) a = fmaf(o[(size_t)t * 128 + k], w2[tid * 128 + k], a);
    sl[tid] = a;
    dout[t * 6 + tid] = a;
  }
  __syncthreads();
  if (tid == 0) {
    float mx = sl[0];
    for (int a = 1; a < 6; ++a) mx = fmaxf(mx, sl[a]);
    float se = 0.0f;
    for (int a = 0; a < 6; ++a) se += __expf(sl[a] - mx);
    float lse = mx + logf(se);
    float ent = 0.0f;
    for (int a = 0; a < 6; ++a) {
      float lp = sl[a] - lse;
      ent -= __expf(lp) * lp;
    }
    dout[3072 + t] = ent;
  }
}

// ------------------------------------------------------------------
// Critic head: pool[128] -> LSTM(h0=c0=0) -> relu lin1 -> lin3 -> v
// ------------------------------------------------------------------
__global__ void k_critic_head(const float* __restrict__ qstar,
                              const float* __restrict__ wih, const float* __restrict__ bih,
                              const float* __restrict__ bhh,
                              const float* __restrict__ l1w, const float* __restrict__ l1b,
                              const float* __restrict__ l3w, const float* __restrict__ l3b,
                              float* __restrict__ dout) {
  __shared__ float sg[512];
  __shared__ float sh2[128];
  __shared__ float so[64];
  int t = threadIdx.x;  // 512
  float g = bih[t] + bhh[t];
  for (int k = 0; k < 128; ++k) g = fmaf(qstar[k], wih[t * 128 + k], g);
  sg[t] = g;
  __syncthreads();
  if (t < 128) {
    float gi = sg[t], gg = sg[256 + t], go = sg[384 + t];  // c0 = 0
    float c2 = sigm(gi) * tanhf(gg);
    sh2[t] = sigm(go) * tanhf(c2);
  }
  __syncthreads();
  if (t < 64) {
    float a = l1b[t];
    for (int k = 0; k < 128; ++k) a = fmaf(sh2[k], l1w[t * 128 + k], a);
    so[t] = fmaxf(a, 0.0f);
  }
  __syncthreads();
  if (t == 0) {
    float v = l3b[0];
    for (int k = 0; k < 64; ++k) v = fmaf(so[k], l3w[k], v);
    dout[3584] = v;
  }
}

// ------------------------------------------------------------------
// Host orchestration
// ------------------------------------------------------------------
struct TrunkP {
  const float *lin0_w, *lin0_b, *mlp_w1, *mlp_b1, *mlp_w2, *mlp_b2, *conv_root, *conv_bias,
              *gru_wih, *gru_whh, *gru_bih, *gru_bhh, *s2s_wih, *s2s_whh, *s2s_bih, *s2s_bhh;
};

struct Bufs {
  bf16 *ews, *h1b, *outB, *mB, *w2b, *wihb, *whhb, *crtb, *memwihb, *lin1wb, *zb, *hactb;
  float *outF, *aggF, *rootF, *giF, *ghF, *degF, *dots, *biasComb, *gact, *oact,
        *qstar, *sh, *sc, *scal, *rraw;
};

inline char* bump(char*& p, size_t bytes) {
  char* r = p;
  p += (bytes + 255) & ~size_t(255);
  return r;
}
inline int ceil_div(long long a, long long b) { return (int)((a + b - 1) / b); }

static void gemm(hipStream_t s, const bf16* A, int lda, const bf16* Bt, int ldb,
                 float* Cf, bf16* Cb, int ldc, const float* bias,
                 int M, int N, int K, int relu) {
  int tiles = (M / 16) * (N / 16);
  k_wmma_gemm<<<ceil_div(tiles, 8), 256, 0, s>>>(A, lda, Bt, ldb, Cf, Cb, ldc, bias,
                                                 M / 16, N / 16, K, relu);
}

static void run_trunk(hipStream_t s, const Bufs& B, const TrunkP& P, const float* x,
                      const float* ea, const int* src, const int* dst) {
  // weight conversions (per trunk, buffers reused)
  k_f32_to_bf16<<<ceil_div(4096 * 128, 256), 256, 0, s>>>(P.mlp_w2, B.w2b, 4096 * 128);
  k_f32_to_bf16<<<ceil_div(192 * 64, 256), 256, 0, s>>>(P.gru_wih, B.wihb, 192 * 64);
  k_f32_to_bf16<<<ceil_div(192 * 64, 256), 256, 0, s>>>(P.gru_whh, B.whhb, 192 * 64);
  k_transpose64_bf16<<<16, 256, 0, s>>>(P.conv_root, B.crtb);
  // node init + edge network
  k_lin0<<<ceil_div(kNodes * 64, 256), 256, 0, s>>>(x, P.lin0_w, P.lin0_b, B.outF);
  k_edge_h1<<<ceil_div(kEdges * 128, 256), 256, 0, s>>>(ea, P.mlp_w1, P.mlp_b1, B.h1b);
  // ew[E,4096] = h1 @ mlp_w2.T + b2  (bf16 NT out, the big WMMA GEMM: 42 GFLOP)
  gemm(s, B.h1b, 128, B.w2b, 128, nullptr, B.ews, 4096, P.mlp_b2, kEdges, 4096, 128, 0);
  // 6 message-passing steps
  for (int step = 0; step < 6; ++step) {
    k_f32_to_bf16<<<ceil_div(kNodes * 64, 256), 256, 0, s>>>(B.outF, B.outB, kNodes * 64);
    hipMemsetAsync(B.aggF, 0, (size_t)kNodes * 64 * sizeof(float), s);
    k_edge_msg<<<ceil_div(kEdges, 8), 256, 0, s>>>(B.outF, B.ews, src, dst, B.aggF);
    gemm(s, B.outB, 64, B.crtb, 64, B.rootF, nullptr, 64, nullptr, kNodes, 64, 64, 0);
    k_m_point<<<ceil_div(kNodes * 64, 256), 256, 0, s>>>(B.aggF, B.degF, B.rootF,
                                                         P.conv_bias, B.mB);
    gemm(s, B.mB, 64, B.wihb, 64, B.giF, nullptr, 192, P.gru_bih, kNodes, 192, 64, 0);
    gemm(s, B.outB, 64, B.whhb, 64, B.ghF, nullptr, 192, P.gru_bhh, kNodes, 192, 64, 0);
    k_gru_point<<<ceil_div(kNodes * 64, 256), 256, 0, s>>>(B.giF, B.ghF, B.outF);
  }
  // Set2Set (6 steps)
  hipMemsetAsync(B.qstar, 0, 128 * sizeof(float), s);
  hipMemsetAsync(B.sh, 0, 64 * sizeof(float), s);
  hipMemsetAsync(B.sc, 0, 64 * sizeof(float), s);
  for (int it = 0; it < 6; ++it) {
    k_s2s_lstm<<<1, 256, 0, s>>>(B.qstar, B.sh, B.sc, P.s2s_wih, P.s2s_whh,
                                 P.s2s_bih, P.s2s_bhh);
    k_s2s_init<<<1, 64, 0, s>>>(B.scal, B.rraw);
    k_dots<<<ceil_div(kNodes, 8), 256, 0, s>>>(B.outF, B.sh, B.dots);
    k_maxred<<<80, 256, 0, s>>>(B.dots, B.scal);
    k_attn<<<ceil_div(kNodes, 8), 256, 0, s>>>(B.dots, B.outF, B.scal, B.rraw);
    k_qstar<<<1, 64, 0, s>>>(B.rraw, B.scal, B.sh, B.qstar);
  }
}

}  // namespace

extern "C" void kernel_launch(void* const* d_in, const int* in_sizes, int n_in,
                              void* d_out, int out_size, void* d_ws, size_t ws_size,
                              hipStream_t stream) {
  (void)in_sizes; (void)n_in; (void)out_size;
  const float* x        = (const float*)d_in[0];
  const int*   eidx     = (const int*)d_in[1];
  const float* eattr    = (const float*)d_in[2];
  const int*   nonring  = (const int*)d_in[3];
  const float* const* A = (const float* const*)(d_in + 4);    // actor params (24)
  const float* const* C = (const float* const*)(d_in + 28);   // critic params (24)
  float* dout = (float*)d_out;
  const int* src = eidx;
  const int* dst = eidx + kEdges;

  TrunkP ta = {A[0], A[1], A[2], A[3], A[4], A[5], A[6], A[7],
               A[8], A[9], A[10], A[11], A[12], A[13], A[14], A[15]};
  TrunkP tc = {C[0], C[1], C[2], C[3], C[4], C[5], C[6], C[7],
               C[8], C[9], C[10], C[11], C[12], C[13], C[14], C[15]};
  const float *mem_wih_a = A[16], *mem_bih_a = A[18], *mem_bhh_a = A[19];
  const float *lin1_w_a = A[20], *lin1_b_a = A[21], *lin2_w_a = A[22], *lin2_b_a = A[23];
  const float *mem_wih_c = C[16], *mem_bih_c = C[18], *mem_bhh_c = C[19];
  const float *lin1_w_c = C[20], *lin1_b_c = C[21], *lin3_w_c = C[22], *lin3_b_c = C[23];
  (void)mem_wih_c;

  char* p = (char*)d_ws;
  char* p0 = p;
  Bufs B;
  B.ews      = (bf16*)bump(p, (size_t)kEdges * 4096 * sizeof(bf16));   // 327.7 MB
  B.h1b      = (bf16*)bump(p, (size_t)kEdges * 128 * sizeof(bf16));
  B.outF     = (float*)bump(p, (size_t)kNodes * 64 * sizeof(float));
  B.outB     = (bf16*)bump(p, (size_t)kNodes * 64 * sizeof(bf16));
  B.aggF     = (float*)bump(p, (size_t)kNodes * 64 * sizeof(float));
  B.rootF    = (float*)bump(p, (size_t)kNodes * 64 * sizeof(float));
  B.mB       = (bf16*)bump(p, (size_t)kNodes * 64 * sizeof(bf16));
  B.giF      = (float*)bump(p, (size_t)kNodes * 192 * sizeof(float));
  B.ghF      = (float*)bump(p, (size_t)kNodes * 192 * sizeof(float));
  B.degF     = (float*)bump(p, (size_t)kNodes * sizeof(float));
  B.dots     = (float*)bump(p, (size_t)kNodes * sizeof(float));
  B.w2b      = (bf16*)bump(p, (size_t)4096 * 128 * sizeof(bf16));
  B.wihb     = (bf16*)bump(p, (size_t)192 * 64 * sizeof(bf16));
  B.whhb     = (bf16*)bump(p, (size_t)192 * 64 * sizeof(bf16));
  B.crtb     = (bf16*)bump(p, (size_t)64 * 64 * sizeof(bf16));
  B.memwihb  = (bf16*)bump(p, (size_t)1536 * 384 * sizeof(bf16));
  B.lin1wb   = (bf16*)bump(p, (size_t)128 * 384 * sizeof(bf16));
  B.zb       = (bf16*)bump(p, (size_t)kTor * 384 * sizeof(bf16));
  B.hactb    = (bf16*)bump(p, (size_t)kTor * 384 * sizeof(bf16));
  B.biasComb = (float*)bump(p, 1536 * sizeof(float));
  B.gact     = (float*)bump(p, (size_t)kTor * 1536 * sizeof(float));
  B.oact     = (float*)bump(p, (size_t)kTor * 128 * sizeof(float));
  B.qstar    = (float*)bump(p, 128 * sizeof(float));
  B.sh       = (float*)bump(p, 64 * sizeof(float));
  B.sc       = (float*)bump(p, 64 * sizeof(float));
  B.scal     = (float*)bump(p, 8 * sizeof(float));
  B.rraw     = (float*)bump(p, 64 * sizeof(float));
  if ((size_t)(p - p0) > ws_size) return;  // workspace too small: bail (compile-only env)

  // degrees (same graph for both trunks)
  hipMemsetAsync(B.degF, 0, (size_t)kNodes * sizeof(float), stream);
  k_deg<<<ceil_div(kEdges, 256), 256, 0, stream>>>(dst, B.degF, kEdges);

  // ---------------- actor ----------------
  run_trunk(stream, B, ta, x, eattr, src, dst);
  k_f32_to_bf16<<<ceil_div(1536 * 384, 256), 256, 0, stream>>>(mem_wih_a, B.memwihb, 1536 * 384);
  k_f32_to_bf16<<<ceil_div(128 * 384, 256), 256, 0, stream>>>(lin1_w_a, B.lin1wb, 128 * 384);
  k_vadd<<<ceil_div(1536, 256), 256, 0, stream>>>(mem_bih_a, mem_bhh_a, B.biasComb, 1536);
  k_build_z<<<ceil_div(kTor * 384, 256), 256, 0, stream>>>(B.outF, nonring, B.qstar, B.zb);
  gemm(stream, B.zb, 384, B.memwihb, 384, B.gact, nullptr, 1536, B.biasComb,
       kTor, 1536, 384, 0);
  k_lstm_act<<<ceil_div(kTor * 384, 256), 256, 0, stream>>>(B.gact, B.hactb);
  gemm(stream, B.hactb, 384, B.lin1wb, 384, B.oact, nullptr, 128, lin1_b_a,
       kTor, 128, 384, 1);
  k_logits_ent<<<kTor, 64, 0, stream>>>(B.oact, lin2_w_a, lin2_b_a, dout);

  // ---------------- critic ----------------
  run_trunk(stream, B, tc, x, eattr, src, dst);
  k_critic_head<<<1, 512, 0, stream>>>(B.qstar, C[16], mem_bih_c, mem_bhh_c,
                                       lin1_w_c, lin1_b_c, lin3_w_c, lin3_b_c, dout);
}
